// UniGS_loss_88252987998645
// MI455X (gfx1250) — compile-verified, matrix operand order checked by reading, not run on hardware
//
#include <hip/hip_runtime.h>

#define B_ 8192
#define D_ 512
#define LDSS 40          // padded LDS row stride in bf16 elems: 32 data + 8 pad (64B + 16B)
#define PANEL (128 * LDSS)

typedef __attribute__((ext_vector_type(16))) __bf16 v16bf;
typedef __attribute__((ext_vector_type(8)))  __bf16 v8bf;
typedef __attribute__((ext_vector_type(8)))  float  v8f;
typedef __attribute__((ext_vector_type(4)))  unsigned int u32x4;
typedef __attribute__((ext_vector_type(8)))  unsigned int u32x8;

// ---- Tensor Data Mover: DMA one 128x32 bf16 panel (row stride 512 elems) into LDS,
// with hardware padding of 4 DWORDs after every 16 DWORDs (-> 40-elem padded rows).
// D# groups per CDNA5 ISA; 2D tensor => 2 SGPR groups (VADDR2/3 NULL).
__device__ __forceinline__ void tdm_load_panel(const __bf16* gsrc, __bf16* ldst) {
  unsigned long long ga = (unsigned long long)(uintptr_t)gsrc;
  unsigned lds = (unsigned)(uintptr_t)ldst;   // generic->LDS: low 32 bits are LDS byte addr

  u32x4 g0;
  g0[0] = 1u;                                  // count=1 (valid user descriptor)
  g0[1] = lds;                                 // lds_addr
  g0[2] = (unsigned)ga;                        // global_addr[31:0]
  g0[3] = (unsigned)(ga >> 32) | 0x80000000u;  // global_addr[56:32] | type=2 ("image")

  u32x8 g1;
  g1[0] = (1u << 16)        // data_size = 1 -> 2 bytes
        | (1u << 20)        // pad_enable
        | (3u << 22)        // pad_interval = 3 -> 16 DWORDs (64B)
        | (3u << 25);       // pad_amount   = 3 -> 4 DWORDs (16B)
  g1[1] = (512u & 0xFFFFu) << 16;              // tensor_dim0[15:0]   (atomic_barrier_addr=0)
  g1[2] = (512u >> 16) | (8192u << 16);        // tensor_dim0[31:16] | tensor_dim1[15:0]
  g1[3] = (8192u >> 16) | (32u << 16);         // tensor_dim1[31:16] | tile_dim0 = 32
  g1[4] = 128u;                                // tile_dim1 = 128, tile_dim2 = 0
  g1[5] = 512u;                                // tensor_dim0_stride[31:0]
  g1[6] = 0u;                                  // stride0[47:32] | stride1[15:0]
  g1[7] = 0u;

  asm volatile("tensor_load_to_lds %0, %1" :: "s"(g0), "s"(g1) : "memory");
}

// A/B fragment for v_wmma_f32_16x16x32_bf16 from row-major padded LDS tile.
// ISA 7.12.2: lanes 0-15 hold row M=lane, K in {0..7, 16..23};
//             lanes 16-31 hold row M=lane-16, K in {8..15, 24..31}.
__device__ __forceinline__ v16bf load_frag(const __bf16* base, int row, int lane) {
  const int kbase = (lane >> 4) << 3;           // 0 or 8
  const __bf16* p = base + row * LDSS + kbase;  // 16B-aligned (80B row stride, 16B k-offsets)
  union { v16bf v; v8bf h[2]; } u;
  u.h[0] = *(const v8bf*)(p);        // K = kbase .. kbase+7
  u.h[1] = *(const v8bf*)(p + 16);   // K = kbase+16 .. kbase+23
  return u.v;
}

__global__ void zero_kernel(float* p, int n) {
  int i = blockIdx.x * blockDim.x + threadIdx.x;
  if (i < n) p[i] = 0.0f;
}

// f32 -> bf16 hi/lo split (hi = RNE(x), lo = RNE(x - hi)) for split-precision GEMM
__global__ void split_bf16_kernel(const float* __restrict__ x,
                                  __bf16* __restrict__ hi,
                                  __bf16* __restrict__ lo, int n) {
  int i = blockIdx.x * blockDim.x + threadIdx.x;
  if (i < n) {
    float v = x[i];
    __bf16 h = (__bf16)v;
    hi[i] = h;
    lo[i] = (__bf16)(v - (float)h);
  }
}

// one wave32 per row: diag_img[i] = s*G_i.I_i ; diag_txt[i] = mask[i,i] ? 0 : s*G_i.T_i
__global__ void diag_kernel(const float* __restrict__ G, const float* __restrict__ I,
                            const float* __restrict__ T,
                            const unsigned char* __restrict__ mask,
                            const float* __restrict__ ls,
                            float* __restrict__ diag_img, float* __restrict__ diag_txt) {
  int gid  = blockIdx.x * blockDim.x + threadIdx.x;
  int row  = gid >> 5;
  int lane = gid & 31;
  const float* g  = G + (size_t)row * D_;
  const float* im = I + (size_t)row * D_;
  const float* tx = T + (size_t)row * D_;
  float di = 0.f, dt = 0.f;
  for (int k = lane; k < D_; k += 32) {
    float gv = g[k];
    di += gv * im[k];
    dt += gv * tx[k];
  }
  for (int off = 16; off > 0; off >>= 1) {
    di += __shfl_xor(di, off, 32);
    dt += __shfl_xor(dt, off, 32);
  }
  if (lane == 0) {
    float s = __expf(ls[0]);
    diag_img[row] = s * di;
    diag_txt[row] = mask[(size_t)row * B_ + row] ? 0.f : s * dt;
  }
}

// 128x128 block tile of A = s*(G @ X^T); 8 waves, each 2x4 WMMA 16x16 tiles.
// Panels stream in via TDM (double-buffered, TENSORcnt-tracked), fully overlapped
// with the 24 v_wmma_f32_16x16x32_bf16 per wave per K-step (split-precision
// acc += hi*hi + hi*lo + lo*hi). Epilogue: exp + (optionally masked) row/col sums.
template <bool MASKED>
__global__ void __launch_bounds__(256)
pair_gemm_reduce(const __bf16* __restrict__ Ahi, const __bf16* __restrict__ Alo,
                 const __bf16* __restrict__ Bhi, const __bf16* __restrict__ Blo,
                 const unsigned char* __restrict__ mask, const float* __restrict__ ls,
                 float* __restrict__ rows_out, float* __restrict__ cols_out) {
  // [buffer][panel: 0=Ahi 1=Alo 2=Bhi 3=Blo][128 rows x 40 elems]
  __shared__ __align__(16) __bf16 sP[2][4][PANEL];
  __shared__ float lds_rows[128];
  __shared__ float lds_cols[128];

  const int tid     = threadIdx.x;
  const int lane    = tid & 31;
  const int w       = tid >> 5;
  const int waveRow = w & 3;   // 4 row groups of 32
  const int waveCol = w >> 2;  // 2 col groups of 64
  const int rowBase = blockIdx.y * 128;
  const int colBase = blockIdx.x * 128;
  const float s = __expf(ls[0]);

  if (tid < 128) { lds_rows[tid] = 0.f; lds_cols[tid] = 0.f; }

  const v8f vzero = {0.f, 0.f, 0.f, 0.f, 0.f, 0.f, 0.f, 0.f};
  v8f acc[2][4];
#pragma unroll
  for (int r = 0; r < 2; ++r)
#pragma unroll
    for (int c = 0; c < 4; ++c) acc[r][c] = vzero;

  const size_t oa0 = (size_t)rowBase * D_;
  const size_t ob0 = (size_t)colBase * D_;

  // prime buffer 0 with the first K panel (one wave issues; EXEC-independent DMA)
  if (w == 0) {
    tdm_load_panel(Ahi + oa0, sP[0][0]);
    tdm_load_panel(Alo + oa0, sP[0][1]);
    tdm_load_panel(Bhi + ob0, sP[0][2]);
    tdm_load_panel(Blo + ob0, sP[0][3]);
  }

  int buf = 0;
  for (int k0 = 0; k0 < D_; k0 += 32) {
    if (w == 0) __builtin_amdgcn_s_wait_tensorcnt(0);  // current buffer's DMA done
    __syncthreads();                                   // publish LDS panel to all waves
    if (w == 0 && k0 + 32 < D_) {                      // launch next panel into other buffer
      tdm_load_panel(Ahi + oa0 + k0 + 32, sP[buf ^ 1][0]);
      tdm_load_panel(Alo + oa0 + k0 + 32, sP[buf ^ 1][1]);
      tdm_load_panel(Bhi + ob0 + k0 + 32, sP[buf ^ 1][2]);
      tdm_load_panel(Blo + ob0 + k0 + 32, sP[buf ^ 1][3]);
    }

    const __bf16* cAhi = sP[buf][0];
    const __bf16* cAlo = sP[buf][1];
    const __bf16* cBhi = sP[buf][2];
    const __bf16* cBlo = sP[buf][3];

    v16bf afh[2], afl[2];
#pragma unroll
    for (int rt = 0; rt < 2; ++rt) {
      int row = waveRow * 32 + rt * 16 + (lane & 15);
      afh[rt] = load_frag(cAhi, row, lane);
      afl[rt] = load_frag(cAlo, row, lane);
    }
#pragma unroll
    for (int ct = 0; ct < 4; ++ct) {
      int col  = waveCol * 64 + ct * 16 + (lane & 15);
      v16bf bfh = load_frag(cBhi, col, lane);
      v16bf bfl = load_frag(cBlo, col, lane);
#pragma unroll
      for (int rt = 0; rt < 2; ++rt) {
        acc[rt][ct] = __builtin_amdgcn_wmma_f32_16x16x32_bf16(
            false, afh[rt], false, bfh, (short)0, acc[rt][ct], false, false);
        acc[rt][ct] = __builtin_amdgcn_wmma_f32_16x16x32_bf16(
            false, afh[rt], false, bfl, (short)0, acc[rt][ct], false, false);
        acc[rt][ct] = __builtin_amdgcn_wmma_f32_16x16x32_bf16(
            false, afl[rt], false, bfh, (short)0, acc[rt][ct], false, false);
      }
    }
    __syncthreads();   // all reads of sP[buf] done before it is overwritten next round
    buf ^= 1;
  }

  // epilogue: exp + (optionally masked) row/col partial sums.
  // C layout (ISA 7.12.2): VGPR v -> M = v + 8*(lane>=16); N = lane&15.
#pragma unroll
  for (int rt = 0; rt < 2; ++rt) {
#pragma unroll
    for (int ct = 0; ct < 4; ++ct) {
      int n_local = waveCol * 64 + ct * 16 + (lane & 15);
      size_t gn = (size_t)(colBase + n_local);
      float csum = 0.f;
#pragma unroll
      for (int v = 0; v < 8; ++v) {
        int m_local = waveRow * 32 + rt * 16 + v + ((lane >> 4) << 3);
        size_t gm = (size_t)(rowBase + m_local);
        float e = __expf(s * acc[rt][ct][v]);
        float rc = e, cv = e;
        if (MASKED) {
          // rows of pc_text: masked iff text_mask[j,i] ; cols (text_pc): iff text_mask[i,j]
          if (mask[gn * B_ + gm]) rc = 1.f;
          if (mask[gm * B_ + gn]) cv = 1.f;
        }
        atomicAdd(&lds_rows[m_local], rc);
        csum += cv;
      }
      atomicAdd(&lds_cols[n_local], csum);
    }
  }
  __syncthreads();
  if (tid < 128) {
    atomicAdd(&rows_out[rowBase + tid], lds_rows[tid]);
    atomicAdd(&cols_out[colBase + tid], lds_cols[tid]);
  }
}

__global__ void finalize_kernel(const float* __restrict__ rows_i, const float* __restrict__ cols_i,
                                const float* __restrict__ rows_t, const float* __restrict__ cols_t,
                                const float* __restrict__ diag_i, const float* __restrict__ diag_t,
                                float* __restrict__ out) {
  __shared__ float red[6];
  if (threadIdx.x < 6) red[threadIdx.x] = 0.f;
  __syncthreads();
  float a0 = 0, a1 = 0, a2 = 0, a3 = 0, a4 = 0, a5 = 0;
  for (int i = threadIdx.x; i < B_; i += blockDim.x) {
    a0 += __logf(rows_i[i]);   // lse rows of pc_image  (no max shift: |logit| <= s ~ 14.29)
    a1 += __logf(cols_i[i]);   // lse rows of image_pc
    a2 += __logf(rows_t[i]);   // lse rows of pc_text
    a3 += __logf(cols_t[i]);   // lse rows of text_pc
    a4 += diag_i[i];
    a5 += diag_t[i];
  }
  atomicAdd(&red[0], a0); atomicAdd(&red[1], a1); atomicAdd(&red[2], a2);
  atomicAdd(&red[3], a3); atomicAdd(&red[4], a4); atomicAdd(&red[5], a5);
  __syncthreads();
  if (threadIdx.x == 0) {
    const float invB = 1.0f / (float)B_;
    float ce_pi = (red[0] - red[4]) * invB;
    float ce_ip = (red[1] - red[4]) * invB;
    float ce_pt = (red[2] - red[5]) * invB;
    float ce_tp = (red[3] - red[5]) * invB;
    float image_3d = 0.25f * (ce_pi + ce_ip);  // 0.5 avg * 0.5 weight
    float text_3d  = 0.25f * (ce_pt + ce_tp);
    out[0] = text_3d + image_3d;
    out[1] = text_3d;
    out[2] = image_3d;
  }
}

extern "C" void kernel_launch(void* const* d_in, const int* in_sizes, int n_in,
                              void* d_out, int out_size, void* d_ws, size_t ws_size,
                              hipStream_t stream) {
  (void)in_sizes; (void)n_in; (void)out_size; (void)ws_size;
  const float* img = (const float*)d_in[0];          // image_features  [B,D]
  const float* txt = (const float*)d_in[1];          // text_features   [B,D]
  const float* gau = (const float*)d_in[2];          // gaussian_features [B,D]
  const unsigned char* mask = (const unsigned char*)d_in[3];  // text_mask [B,B] bool
  const float* ls = (const float*)d_in[5];           // logit_scale (scalar)
  float* out = (float*)d_out;                        // [loss, text_3d, image_3d]

  const size_t S = (size_t)B_ * D_;
  __bf16* Ghi = (__bf16*)d_ws;
  __bf16* Glo = Ghi + S;
  __bf16* Ihi = Glo + S;
  __bf16* Ilo = Ihi + S;
  __bf16* Thi = Ilo + S;
  __bf16* Tlo = Thi + S;
  float* rows_img = (float*)(Tlo + S);
  float* cols_img = rows_img + B_;
  float* rows_txt = cols_img + B_;
  float* cols_txt = rows_txt + B_;
  float* diag_img = cols_txt + B_;
  float* diag_txt = diag_img + B_;

  const int n = (int)S;  // 4,194,304
  split_bf16_kernel<<<n / 256, 256, 0, stream>>>(gau, Ghi, Glo, n);
  split_bf16_kernel<<<n / 256, 256, 0, stream>>>(img, Ihi, Ilo, n);
  split_bf16_kernel<<<n / 256, 256, 0, stream>>>(txt, Thi, Tlo, n);
  zero_kernel<<<(4 * B_) / 256, 256, 0, stream>>>(rows_img, 4 * B_);
  diag_kernel<<<(B_ * 32) / 256, 256, 0, stream>>>(gau, img, txt, mask, ls, diag_img, diag_txt);

  dim3 grid(B_ / 128, B_ / 128);  // 64 x 64 tiles
  pair_gemm_reduce<false><<<grid, 256, 0, stream>>>(Ghi, Glo, Ihi, Ilo, nullptr, ls,
                                                    rows_img, cols_img);
  pair_gemm_reduce<true><<<grid, 256, 0, stream>>>(Ghi, Glo, Thi, Tlo, mask, ls,
                                                   rows_txt, cols_txt);

  finalize_kernel<<<1, 256, 0, stream>>>(rows_img, cols_img, rows_txt, cols_txt,
                                         diag_img, diag_txt, out);
}
